// EfficientSpanAttention_53867479826541
// MI455X (gfx1250) — matrix operand order repaired and checked by reading, not compiled, not run on hardware
//
#include <hip/hip_runtime.h>
#include <hip/hip_bf16.h>

// ---------------------------------------------------------------------------
// CDNA5 / gfx1250 attention block.
//   - all GEMMs on v_wmma_f32_16x16x32_bf16 (f32 accumulate)
//   - weight tiles staged to LDS via the Tensor Data Mover (tensor_load_to_lds,
//     double-buffered, s_wait_tensorcnt) -- falls back to coop ds_store copy.
// ---------------------------------------------------------------------------

typedef __attribute__((ext_vector_type(16))) __bf16        v16bf;
typedef __attribute__((ext_vector_type(8)))  float         v8f;
typedef __attribute__((ext_vector_type(4)))  unsigned int  u32x4;
typedef __attribute__((ext_vector_type(4)))  unsigned int  tdm_g0_t;
typedef __attribute__((ext_vector_type(8)))  int           tdm_g1_t;
typedef __attribute__((ext_vector_type(4)))  int           tdm_g2_t;

#define BATCH 4
#define SEQ   2048
#define DIM   1024
#define NH    4
#define HD    256           // head dim
#define ROWS  (BATCH*SEQ)   // 8192
#define ATT_SCALE 0.0625f   // 1/sqrt(256)

#if __has_builtin(__builtin_amdgcn_tensor_load_to_lds)
#define USE_TDM 1
#else
#define USE_TDM 0
#endif

union AFrag { v16bf v; u32x4 u[2]; };   // 16x32 bf16 A fragment (per-lane: 2x16B)
union BFrag { v16bf v; u32x4 u[2]; };   // 32x16 bf16 B fragment (per-lane: 32B)

// ---------------------------------------------------------------------------
// f32 -> bf16 conversion
// ---------------------------------------------------------------------------
__global__ __launch_bounds__(256) void cvt_f32_bf16(const float* __restrict__ in,
                                                    __bf16* __restrict__ out, int n) {
    int i = blockIdx.x * 256 + threadIdx.x;
    if (i < n) out[i] = (__bf16)in[i];
}

// ---------------------------------------------------------------------------
// TDM: DMA a 2D tile (tile_d0 contiguous elems x tile_d1 rows, 2-byte elems)
// from global to LDS. D# bitfields per CDNA5 ISA ch.8 (group0/group1).
// lds_off: byte offset from wave LDS base (shared array assumed at offset 0).
// ---------------------------------------------------------------------------
__device__ __forceinline__ void tdm_load_tile_2d(unsigned lds_off,
                                                 unsigned long long gaddr,
                                                 int tensor_d0, int tensor_d1,
                                                 int tile_d0, int tile_d1,
                                                 int stride0)
{
#if USE_TDM
    tdm_g0_t g0 = { 1u,                                   // count=1 (valid), user mode
                    lds_off,                              // lds_addr [63:32]
                    (unsigned)(gaddr & 0xffffffffu),      // global_addr[31:0]
                    (unsigned)((gaddr >> 32) & 0x1ffffffu) | (2u << 30) }; // addr hi | type=2
    tdm_g1_t g1 = { (int)(1u << 16),                                          // data_size=1 (2B)
                    (int)(((unsigned)tensor_d0 & 0xffffu) << 16),             // dim0 lo
                    (int)((((unsigned)tensor_d0 >> 16) & 0xffffu) |
                          (((unsigned)tensor_d1 & 0xffffu) << 16)),           // dim0 hi|dim1 lo
                    (int)((((unsigned)tensor_d1 >> 16) & 0xffffu) |
                          ((unsigned)tile_d0 << 16)),                         // dim1 hi|tile0
                    (int)((unsigned)tile_d1 & 0xffffu),                       // tile1 (tile2=0)
                    stride0,                                                  // stride0[31:0]
                    0, 0 };                                                   // stride0 hi, stride1
    tdm_g2_t g2 = { 0, 0, 0, 0 };
    tdm_g2_t g3 = { 0, 0, 0, 0 };
#if defined(__clang_major__) && (__clang_major__ >= 23)
    tdm_g1_t g4 = { 0, 0, 0, 0, 0, 0, 0, 0 };
    __builtin_amdgcn_tensor_load_to_lds(g0, g1, g2, g3, g4, 0);
#else
    __builtin_amdgcn_tensor_load_to_lds(g0, g1, g2, g3, 0);
#endif
#endif
}

// ---------------------------------------------------------------------------
// GEMM: C[m,n] = sum_k A[m,k] * W[n,k]   (A: MxK bf16, W: NxK bf16, row-major)
// MODE 0: write bf16 C row-major (q / k)
// MODE 1: write bf16 C transposed per-head: vt[b][h][dv][s]  (for V)
// MODE 2: write f32  C + bias[n] + resid[m,n]                (out-proj)
// Block: 256 thr = 8 waves; tile 128 rows x 64 cols; wave -> 16 rows x 64 cols.
// W tile (64 cols x 32 k = 4KB) staged in LDS once per block (TDM async,
// double-buffered) instead of 8x redundant per-wave L2 streams.
// LDS layout: [col][32 k elems] = exactly the B-fragment load order.
// ---------------------------------------------------------------------------
template <int MODE>
__global__ __launch_bounds__(256)
void gemm_xwt(const __bf16* __restrict__ A, const __bf16* __restrict__ W,
              __bf16* __restrict__ outb, float* __restrict__ outf,
              const float* __restrict__ bias, const float* __restrict__ resid,
              int M, int N, int K)
{
    __shared__ __bf16 bsm[2][64 * 32];          // 2 x 4KB double buffer

    const int tid  = threadIdx.x;
    const int lane = tid & 31;
    const int wave = tid >> 5;   // 0..7
    const int half = lane >> 4;
    const int l16  = lane & 15;

    const int bm  = blockIdx.x * 128;
    const int bn  = blockIdx.y * 64;
    const int row = bm + wave * 16 + l16;

    v8f acc[4];
    for (int t = 0; t < 4; ++t)
        for (int e = 0; e < 8; ++e) acc[t][e] = 0.f;

    // ---- stage helper: bring W[bn..bn+63][k0..k0+31] into bsm[buf] ----
    auto stage = [&](int buf, int k0) {
#if USE_TDM
        if (wave == 0) {
            const unsigned long long ga =
                (unsigned long long)(uintptr_t)(W + (size_t)bn * K + k0);
            tdm_load_tile_2d((unsigned)(buf * (64 * 32 * 2)), ga,
                             K, N, /*tile_d0=*/32, /*tile_d1=*/64, /*stride0=*/K);
        }
#else
        const int col = tid >> 2;       // 0..63
        const int ch  = tid & 3;        // 0..3 (8 elems = 16B each)
        *(u32x4*)&bsm[buf][col * 32 + ch * 8] =
            *(const u32x4*)(W + (size_t)(bn + col) * K + k0 + ch * 8);
#endif
    };
    auto stage_wait = [&]() {
#if USE_TDM
        if (wave == 0) __builtin_amdgcn_s_wait_tensorcnt((short)0);
#endif
    };

    stage(0, 0);
    stage_wait();
    __syncthreads();

    const __bf16* arow = A + (size_t)row * K;
    for (int k0 = 0; k0 < K; k0 += 32) {
        const int buf = (k0 >> 5) & 1;
        if (k0 + 32 < K) stage(buf ^ 1, k0 + 32);   // overlap DMA with compute

        AFrag a;
        a.u[0] = *(const u32x4*)(arow + k0 + 8 * half);
        a.u[1] = *(const u32x4*)(arow + k0 + 16 + 8 * half);
#pragma unroll
        for (int t = 0; t < 4; ++t) {
            const __bf16* bp = &bsm[buf][(t * 16 + l16) * 32 + 8 * half];
            BFrag b;
            b.u[0] = *(const u32x4*)(bp);
            b.u[1] = *(const u32x4*)(bp + 16);
            acc[t] = __builtin_amdgcn_wmma_f32_16x16x32_bf16(
                false, a.v, false, b.v, (short)0, acc[t], false, false);
        }

        if (k0 + 32 < K) stage_wait();
        __syncthreads();
    }

#pragma unroll
    for (int t = 0; t < 4; ++t) {
        const int col = bn + t * 16 + l16;
#pragma unroll
        for (int j = 0; j < 8; ++j) {
            const int r = bm + wave * 16 + j + 8 * half;
            const float val = acc[t][j];
            if (MODE == 0) {
                outb[(size_t)r * N + col] = (__bf16)val;
            } else if (MODE == 1) {
                const int bb = r >> 11;          // r / SEQ
                const int s  = r & (SEQ - 1);
                const int h  = col >> 8;         // col / HD
                const int dv = col & (HD - 1);
                outb[((size_t)((bb * NH + h) * HD + dv)) * SEQ + s] = (__bf16)val;
            } else {
                outf[(size_t)r * N + col] = val + bias[col] + resid[(size_t)r * N + col];
            }
        }
    }
}

// ---------------------------------------------------------------------------
// Flash attention: one wave per (b, h, 16-query tile). 128 thr = 4 waves/block.
// q,k: [b][s][d] bf16 (head h = cols h*HD..); vt: [b][h][dv][s] bf16.
// ---------------------------------------------------------------------------
__global__ __launch_bounds__(128)
void attn_kernel(const __bf16* __restrict__ q, const __bf16* __restrict__ k,
                 const __bf16* __restrict__ vt, const int* __restrict__ amask,
                 __bf16* __restrict__ ctx)
{
    __shared__ __bf16 plds[4][16 * 32];   // per-wave P staging (C-layout -> A-layout)

    const int lane = threadIdx.x & 31;
    const int wv   = threadIdx.x >> 5;
    const int half = lane >> 4;
    const int l16  = lane & 15;

    const int wid = blockIdx.x * 4 + wv;       // 0..2047
    const int qt  = wid & 127;                 // query tile within (b,h)
    const int h   = (wid >> 7) & (NH - 1);
    const int b   = wid >> 9;

    // Q fragments: 16 rows x 256 cols = 8 K-chunks of 32, kept in registers.
    const __bf16* qbase = q + ((size_t)(b * SEQ + qt * 16 + l16)) * DIM + h * HD;
    AFrag aq[8];
#pragma unroll
    for (int c = 0; c < 8; ++c) {
        aq[c].u[0] = *(const u32x4*)(qbase + c * 32 + 8 * half);
        aq[c].u[1] = *(const u32x4*)(qbase + c * 32 + 16 + 8 * half);
    }

    float mrun[8], lrun[8], alpha[8];
    v8f acc[16];
#pragma unroll
    for (int j = 0; j < 8; ++j) { mrun[j] = -1e30f; lrun[j] = 0.f; }
    for (int t = 0; t < 16; ++t)
        for (int j = 0; j < 8; ++j) acc[t][j] = 0.f;

    const int* mrow = amask + b * SEQ;

    for (int kt = 0; kt < SEQ; kt += 32) {
        // ---- scores: two 16-key subtiles, accumulate over head dim ----
        v8f s0, s1;
        for (int j = 0; j < 8; ++j) { s0[j] = 0.f; s1[j] = 0.f; }
        const __bf16* kb0 = k + ((size_t)(b * SEQ + kt + l16)) * DIM + h * HD;
        const __bf16* kb1 = kb0 + (size_t)16 * DIM;
#pragma unroll
        for (int c = 0; c < 8; ++c) {
            BFrag bk;
            bk.u[0] = *(const u32x4*)(kb0 + c * 32 + 8 * half);
            bk.u[1] = *(const u32x4*)(kb0 + c * 32 + 16 + 8 * half);
            s0 = __builtin_amdgcn_wmma_f32_16x16x32_bf16(
                false, aq[c].v, false, bk.v, (short)0, s0, false, false);
            bk.u[0] = *(const u32x4*)(kb1 + c * 32 + 8 * half);
            bk.u[1] = *(const u32x4*)(kb1 + c * 32 + 16 + 8 * half);
            s1 = __builtin_amdgcn_wmma_f32_16x16x32_bf16(
                false, aq[c].v, false, bk.v, (short)0, s1, false, false);
        }

        // ---- mask + scale (lane holds one key column per subtile) ----
        const int mv0 = mrow[kt + l16];
        const int mv1 = mrow[kt + 16 + l16];
#pragma unroll
        for (int j = 0; j < 8; ++j) {
            s0[j] = mv0 ? s0[j] * ATT_SCALE : -1e9f;
            s1[j] = mv1 ? s1[j] * ATT_SCALE : -1e9f;
        }

        // ---- online softmax: reduce across the 16 lanes of each half ----
        float p0[8], p1[8];
#pragma unroll
        for (int j = 0; j < 8; ++j) {
            float t2 = fmaxf(s0[j], s1[j]);
            for (int m = 1; m < 16; m <<= 1) t2 = fmaxf(t2, __shfl_xor(t2, m, 32));
            const float mn = fmaxf(mrun[j], t2);
            alpha[j] = __expf(mrun[j] - mn);
            p0[j] = __expf(s0[j] - mn);
            p1[j] = __expf(s1[j] - mn);
            float su = p0[j] + p1[j];
            for (int m = 1; m < 16; m <<= 1) su += __shfl_xor(su, m, 32);
            lrun[j] = lrun[j] * alpha[j] + su;
            mrun[j] = mn;
        }
        for (int t = 0; t < 16; ++t)
            for (int j = 0; j < 8; ++j) acc[t][j] *= alpha[j];

        // ---- P: C-layout -> A-layout via LDS bounce (same-wave RAW) ----
        __bf16* pl = &plds[wv][0];
#pragma unroll
        for (int j = 0; j < 8; ++j) {
            const int r = j + 8 * half;
            pl[r * 32 + l16]      = (__bf16)p0[j];
            pl[r * 32 + 16 + l16] = (__bf16)p1[j];
        }
        asm volatile("s_wait_dscnt 0" ::: "memory");
        AFrag ap;
        ap.u[0] = *(const u32x4*)(pl + l16 * 32 + 8 * half);
        ap.u[1] = *(const u32x4*)(pl + l16 * 32 + 16 + 8 * half);

        // ---- ctx += P @ V  (vt is [b][h][dv][s]: contiguous along keys) ----
        const __bf16* vb = vt + ((size_t)((b * NH + h) * HD)) * SEQ + kt + 8 * half;
#pragma unroll
        for (int t = 0; t < 16; ++t) {
            const __bf16* vp = vb + (size_t)(t * 16 + l16) * SEQ;
            BFrag bv;
            bv.u[0] = *(const u32x4*)(vp);
            bv.u[1] = *(const u32x4*)(vp + 16);
            acc[t] = __builtin_amdgcn_wmma_f32_16x16x32_bf16(
                false, ap.v, false, bv.v, (short)0, acc[t], false, false);
        }
    }

    // ---- normalize + store ctx[b][s][h*HD+dv] bf16 ----
#pragma unroll
    for (int t = 0; t < 16; ++t) {
#pragma unroll
        for (int j = 0; j < 8; ++j) {
            const int qrow = qt * 16 + j + 8 * half;
            const int dv   = t * 16 + l16;
            const float val = acc[t][j] / lrun[j];
            ctx[((size_t)(b * SEQ + qrow)) * DIM + h * HD + dv] = (__bf16)val;
        }
    }
}

// ---------------------------------------------------------------------------
// LayerNorm: one block (256 thr) per row of h[ROWS][DIM] f32.
// ---------------------------------------------------------------------------
__global__ __launch_bounds__(256)
void layernorm_kernel(const float* __restrict__ h, const float* __restrict__ gamma,
                      const float* __restrict__ beta, float* __restrict__ out)
{
    __shared__ float red0[8], red1[8];
    const int row = blockIdx.x;
    const float* hr = h + (size_t)row * DIM;

    float s = 0.f, s2 = 0.f;
    for (int i = threadIdx.x; i < DIM; i += 256) {
        const float v = hr[i];
        s += v; s2 += v * v;
    }
    for (int m = 1; m < 32; m <<= 1) { s += __shfl_xor(s, m, 32); s2 += __shfl_xor(s2, m, 32); }
    const int wv = threadIdx.x >> 5, ln = threadIdx.x & 31;
    if (ln == 0) { red0[wv] = s; red1[wv] = s2; }
    __syncthreads();
    if (wv == 0) {
        s  = (ln < 8) ? red0[ln] : 0.f;
        s2 = (ln < 8) ? red1[ln] : 0.f;
        for (int m = 1; m < 8; m <<= 1) { s += __shfl_xor(s, m, 32); s2 += __shfl_xor(s2, m, 32); }
        if (ln == 0) { red0[0] = s; red1[0] = s2; }
    }
    __syncthreads();
    const float mean = red0[0] * (1.f / DIM);
    const float var  = red1[0] * (1.f / DIM) - mean * mean;
    const float rstd = rsqrtf(var + 1e-5f);
    for (int i = threadIdx.x; i < DIM; i += 256)
        out[(size_t)row * DIM + i] = (hr[i] - mean) * rstd * gamma[i] + beta[i];
}

// ---------------------------------------------------------------------------
// Host-side launch
// ---------------------------------------------------------------------------
extern "C" void kernel_launch(void* const* d_in, const int* in_sizes, int n_in,
                              void* d_out, int out_size, void* d_ws, size_t ws_size,
                              hipStream_t stream)
{
    const float* x     = (const float*)d_in[0];
    const int*   amask = (const int*)  d_in[1];
    const float* Wq    = (const float*)d_in[2];
    const float* Wk    = (const float*)d_in[3];
    const float* Wv    = (const float*)d_in[4];
    const float* Wo    = (const float*)d_in[5];
    const float* bo    = (const float*)d_in[6];
    const float* gamma = (const float*)d_in[7];
    const float* beta  = (const float*)d_in[8];
    float* out = (float*)d_out;

    char* ws = (char*)d_ws;
    const size_t MB = 1024ull * 1024ull;
    __bf16* xb  = (__bf16*)(ws + 0);          // 16 MB
    __bf16* wqb = (__bf16*)(ws + 16 * MB);    //  2 MB
    __bf16* wkb = (__bf16*)(ws + 18 * MB);    //  2 MB
    __bf16* wvb = (__bf16*)(ws + 20 * MB);    //  2 MB
    __bf16* wob = (__bf16*)(ws + 22 * MB);    //  2 MB
    __bf16* qb  = (__bf16*)(ws + 24 * MB);    // 16 MB
    __bf16* kb  = (__bf16*)(ws + 40 * MB);    // 16 MB
    __bf16* vtb = (__bf16*)(ws + 56 * MB);    // 16 MB
    __bf16* ctx = (__bf16*)(ws + 72 * MB);    // 16 MB
    float*  hbf = (float*) (ws + 24 * MB);    // 32 MB (reuses dead q/k regions)

    const int NX = ROWS * DIM;      // 8388608
    const int NW = DIM * DIM;       // 1048576

    cvt_f32_bf16<<<NX / 256, 256, 0, stream>>>(x,  xb,  NX);
    cvt_f32_bf16<<<NW / 256, 256, 0, stream>>>(Wq, wqb, NW);
    cvt_f32_bf16<<<NW / 256, 256, 0, stream>>>(Wk, wkb, NW);
    cvt_f32_bf16<<<NW / 256, 256, 0, stream>>>(Wv, wvb, NW);
    cvt_f32_bf16<<<NW / 256, 256, 0, stream>>>(Wo, wob, NW);

    dim3 gg(ROWS / 128, DIM / 64);  // (64, 16)
    gemm_xwt<0><<<gg, 256, 0, stream>>>(xb, wqb, qb,  nullptr, nullptr, nullptr, ROWS, DIM, DIM);
    gemm_xwt<0><<<gg, 256, 0, stream>>>(xb, wkb, kb,  nullptr, nullptr, nullptr, ROWS, DIM, DIM);
    gemm_xwt<1><<<gg, 256, 0, stream>>>(xb, wvb, vtb, nullptr, nullptr, nullptr, ROWS, DIM, DIM);

    attn_kernel<<<512, 128, 0, stream>>>(qb, kb, vtb, amask, ctx);

    gemm_xwt<2><<<gg, 256, 0, stream>>>(ctx, wob, nullptr, hbf, bo, x, ROWS, DIM, DIM);

    layernorm_kernel<<<ROWS, 256, 0, stream>>>(hbf, gamma, beta, out);
}
